// GPTrack2D_39067022524776
// MI455X (gfx1250) — compile-verified, matrix operand order checked by reading, not run on hardware
//
#include <hip/hip_runtime.h>

// ---------------------------------------------------------------------------
// GPTrack2D encoder for MI455X (gfx1250, wave32, WMMA).
// All GEMMs run on v_wmma_f32_16x16x32_bf16 with fp32 accumulation.
// Double-buffered LDS pipeline; A-tiles via async global->LDS DMA.
// ---------------------------------------------------------------------------

typedef __attribute__((ext_vector_type(16))) __bf16 v16bf;
typedef __attribute__((ext_vector_type(8)))  float  v8f;
typedef int v4i __attribute__((vector_size(16)));   // matches async builtin param

union FragU { unsigned u[8]; v16bf v; };
union U4    { uint4 q; unsigned u[4]; unsigned short s[8]; };

#if __has_builtin(__builtin_amdgcn_global_load_async_to_lds_b128)
#define HAVE_ASYNC_LDS 1
#endif

#define DEV __device__ __forceinline__

static constexpr int  Bc   = 2;
static constexpr int  Lc   = 12;
static constexpr int  Nc   = 1024;
static constexpr int  Dc   = 768;
static constexpr int  Hc   = 12;
static constexpr int  TDc  = 2304;   // 3*D
static constexpr int  Mc   = 3072;   // MLP hidden
static constexpr int  ROWS = Bc * Nc;                 // 2048 GEMM rows
static constexpr long NDl  = (long)Nc * Dc;           // 786432
static constexpr long LNDl = (long)Lc * NDl;
static constexpr long BNDl = (long)Bc * NDl;
static constexpr long BLNDl= (long)Bc * LNDl;

DEV unsigned short f2bf(float f) {
  unsigned u = __float_as_uint(f);
  u += 0x7FFFu + ((u >> 16) & 1u);          // round-to-nearest-even
  return (unsigned short)(u >> 16);
}
DEV float elu1(float x)  { return x > 0.f ? x + 1.f : __expf(x); }
DEV float gelu_f(float x){ return 0.5f * x * (1.f + erff(x * 0.70710678118654752440f)); }

// ---------------------------------------------------------------------------
// General batched bf16 WMMA GEMM:  C(f32 and/or bf16) = act(A*B + bias [+C])
// Block tile 128x64, 8 waves, each wave a 32x32 patch = 2x2 WMMA 16x16x32.
// Requires K % 32 == 0 (all call sites satisfy this).
// Software-pipelined, double-buffered LDS, one barrier per K tile.
// Batch index z: outer = z / hDiv, inner = z % hDiv (two-level strides).
// ---------------------------------------------------------------------------
struct GemmArgs {
  const unsigned short* A;
  const unsigned short* Bm;
  float*          C;
  unsigned short* Cbf;
  const float*    bias;
  int  M, N, K;
  long sAm, sAk, sBk, sBn, sCm, sCbm;
  long aBO, aBI, bBO, bBI, cBO, cBI;
  int  hDiv, accumulate, act;   // act: 0 none, 1 gelu
};

__global__ __launch_bounds__(256)
void gemm_bf16_wmma(GemmArgs g) {
  __shared__ unsigned short lA[2][128 * 32];  // A tile, row-major [m][k]
  __shared__ unsigned short lBT[2][64 * 32];  // B tile transposed [n][k], (k,k+1) packed

  const int tid  = threadIdx.x;
  const int lane = tid & 31;
  const int wave = tid >> 5;
  const int wr   = wave >> 1;               // 0..3 -> 32-row band
  const int wc   = wave & 1;                // 0..1 -> 32-col band
  const int bm   = blockIdx.y * 128;
  const int bn   = blockIdx.x * 64;
  const int zo   = blockIdx.z / g.hDiv;
  const int zi   = blockIdx.z % g.hDiv;

  const unsigned short* A  = g.A  + (long)zo * g.aBO + (long)zi * g.aBI;
  const unsigned short* Bm = g.Bm + (long)zo * g.bBO + (long)zi * g.bBI;
  const long cOff = (long)zo * g.cBO + (long)zi * g.cBI;

  const int half8 = (lane >= 16) ? 8 : 0;
  const int lrow  = lane & 15;

  const bool fastA  = (g.sAk == 1) && (bm + 128 <= g.M);           // row-major A
  const bool fastAT = (g.sAm == 1) && (bm == 0) && (g.M <= 64);    // kv einsum (A^T)
  const bool fastB  = (g.sBn == 1) && (bn + 64  <= g.N);

  const int nT = g.K / 32;
  v8f acc[2][2] = {};
  U4  ra0, ra1, rb0, rb1;     // in-flight registers for pipelined tile

  // zero-fill A padding rows once (fastAT only touches rows 0..63)
  if (fastAT) {
    uint4 z; z.x = z.y = z.z = z.w = 0u;
    *(uint4*)&lA[0][64 * 32 + tid * 8] = z;
    *(uint4*)&lA[1][64 * 32 + tid * 8] = z;
  }

  // ---------------- staging: issue (global side) ----------------
  auto stageA_issue = [&](int k0, int buf) {
    if (fastA) {
      #pragma unroll
      for (int j = 0; j < 2; ++j) {
        int c  = tid * 2 + j;                 // 16B chunk id
        int m  = c >> 2;
        int ko = (c & 3) * 8;
        const unsigned short* src = A + (long)(bm + m) * g.sAm + (k0 + ko);
#ifdef HAVE_ASYNC_LDS
        __builtin_amdgcn_global_load_async_to_lds_b128(
            (__attribute__((address_space(1))) v4i*)(unsigned short*)src,
            (__attribute__((address_space(3))) v4i*)&lA[buf][m * 32 + ko], 0, 0);
#else
        *(uint4*)&lA[buf][m * 32 + ko] = *(const uint4*)src;
#endif
      }
    } else if (fastAT) {
      if (tid < 128) {
        int k  = (tid >> 3) * 2;              // even k pair
        int m0 = (tid & 7) * 8;               // 8 consecutive m
        const unsigned short* a0 = A + (long)(k0 + k) * g.sAk + m0;
        ra0.q = *(const uint4*)a0;
        ra1.q = *(const uint4*)(a0 + g.sAk);
      }
    } else {
      #pragma unroll
      for (int e = 0; e < 16; ++e) {
        int flat = tid * 16 + e;
        int m = flat >> 5, kk = flat & 31;
        int gm = bm + m;
        int cm = gm < g.M ? gm : g.M - 1;     // clamp: no exec-mask churn
        unsigned short v = A[(long)cm * g.sAm + (long)(k0 + kk) * g.sAk];
        if (gm >= g.M) v = 0;
        lA[buf][m * 32 + kk] = v;
      }
    }
  };
  auto stageB_issue = [&](int k0, int buf) {
    if (fastB) {
      if (tid < 128) {
        int k  = (tid >> 3) * 2;
        int n0 = (tid & 7) * 8;
        const unsigned short* b0 = Bm + (long)(k0 + k) * g.sBk + (bn + n0);
        rb0.q = *(const uint4*)b0;
        rb1.q = *(const uint4*)(b0 + g.sBk);
      }
    } else {
      #pragma unroll
      for (int e = 0; e < 8; ++e) {
        int flat = tid * 8 + e;
        int kk = flat >> 6, n = flat & 63;
        int gn = bn + n;
        int cn = gn < g.N ? gn : g.N - 1;
        unsigned short v = Bm[(long)(k0 + kk) * g.sBk + (long)cn * g.sBn];
        if (gn >= g.N) v = 0;
        lBT[buf][n * 32 + kk] = v;
      }
    }
  };
  // ---------------- staging: commit (register -> LDS) ----------------
  auto stage_commit = [&](int buf) {
    if (fastAT && tid < 128) {
      int k  = (tid >> 3) * 2;
      int m0 = (tid & 7) * 8;
      #pragma unroll
      for (int j = 0; j < 8; ++j) {
        unsigned w = (unsigned)ra0.s[j] | ((unsigned)ra1.s[j] << 16);
        *(unsigned*)&lA[buf][(m0 + j) * 32 + k] = w;
      }
    }
    if (fastB && tid < 128) {
      int k  = (tid >> 3) * 2;
      int n0 = (tid & 7) * 8;
      #pragma unroll
      for (int j = 0; j < 8; ++j) {
        unsigned w = (unsigned)rb0.s[j] | ((unsigned)rb1.s[j] << 16);
        *(unsigned*)&lBT[buf][(n0 + j) * 32 + k] = w;
      }
    }
  };

  // prologue: issue tile 0
  stageA_issue(0, 0);
  stageB_issue(0, 0);

  for (int i = 0; i < nT; ++i) {
    const int buf = i & 1;
    stage_commit(buf);
#ifdef HAVE_ASYNC_LDS
    if (fastA) {
  #if __has_builtin(__builtin_amdgcn_s_wait_asynccnt)
      __builtin_amdgcn_s_wait_asynccnt(0);
  #else
      asm volatile("s_wait_asynccnt 0x0" ::: "memory");
  #endif
    }
#endif
    __syncthreads();                        // tile i fully resident
    if (i + 1 < nT) {                       // overlap next tile's global traffic
      stageA_issue((i + 1) * 32, buf ^ 1);
      stageB_issue((i + 1) * 32, buf ^ 1);
    }

    // ---- fragment loads per documented CDNA5 16-bit layouts (merge to b128) ----
    FragU fa[2], fb[2];
    #pragma unroll
    for (int tm = 0; tm < 2; ++tm) {
      int row = wr * 32 + tm * 16 + lrow;
      #pragma unroll
      for (int v = 0; v < 8; ++v) {
        int kk = 2 * (v & 3) + ((v >> 2) << 4) + half8;   // K pair index
        fa[tm].u[v] = *(const unsigned*)&lA[buf][row * 32 + kk];
      }
    }
    #pragma unroll
    for (int tn = 0; tn < 2; ++tn) {
      int row = wc * 32 + tn * 16 + lrow;
      #pragma unroll
      for (int v = 0; v < 8; ++v) {
        int kk = 2 * (v & 3) + ((v >> 2) << 4) + half8;
        fb[tn].u[v] = *(const unsigned*)&lBT[buf][row * 32 + kk];
      }
    }

    #pragma unroll
    for (int tm = 0; tm < 2; ++tm)
      #pragma unroll
      for (int tn = 0; tn < 2; ++tn)
        acc[tm][tn] = __builtin_amdgcn_wmma_f32_16x16x32_bf16(
            false, fa[tm].v, false, fb[tn].v,
            (short)0, acc[tm][tn], false, false);
  }

  // ---- epilogue: C 16x16 f32 layout (VGPR r -> M = r + 8*(lane>=16)) ----
  #pragma unroll
  for (int tm = 0; tm < 2; ++tm) {
    #pragma unroll
    for (int tn = 0; tn < 2; ++tn) {
      #pragma unroll
      for (int r = 0; r < 8; ++r) {
        int m  = wr * 32 + tm * 16 + r + half8;
        int n  = wc * 32 + tn * 16 + lrow;
        int gm = bm + m, gn = bn + n;
        if (gm < g.M && gn < g.N) {
          float v = acc[tm][tn][r];
          if (g.bias) v += g.bias[gn];
          if (g.act == 1) v = gelu_f(v);
          if (g.C) {
            long idx = cOff + (long)gm * g.sCm + gn;
            if (g.accumulate) v += g.C[idx];
            g.C[idx] = v;
          }
          if (g.Cbf) g.Cbf[cOff + (long)gm * g.sCbm + gn] = f2bf(v);
        }
      }
    }
  }
}

// ---------------------------------------------------------------------------
// Elementwise / reduction kernels
// ---------------------------------------------------------------------------

__global__ void f32_to_bf16_kernel(unsigned short* dst, const float* src, long n) {
  long i = (long)blockIdx.x * blockDim.x + threadIdx.x;
  if (i < n) dst[i] = f2bf(src[i]);
}

// xw[b,t,n,d] = xin[b,t,n,d] + tpos[b,t,d] * spos[b,n,d]
__global__ void add_pos_kernel(float* xw, const float* xin,
                               const float* tpos, const float* spos) {
  long i = (long)blockIdx.x * blockDim.x + threadIdx.x;
  if (i >= BLNDl) return;
  int d = (int)(i % Dc);
  long r = i / Dc;
  int n = (int)(r % Nc); r /= Nc;
  int t = (int)(r % Lc);
  int b = (int)(r / Lc);
  xw[i] = xin[i] + tpos[((long)b * Lc + t) * Dc + d] * spos[((long)b * Nc + n) * Dc + d];
}

// hin[b,n,d] = h[b,n,d] + tpos[b,pt,d] * spos[b,n,d]
__global__ void hin_kernel(float* hin, const float* h,
                           const float* tpos, const float* spos, int pt) {
  long i = (long)blockIdx.x * blockDim.x + threadIdx.x;
  if (i >= BNDl) return;
  int d = (int)(i % Dc);
  int b = (int)(i / NDl);
  hin[i] = h[i] + tpos[((long)b * Lc + pt) * Dc + d] * spos[i];
}

// LayerNorm over D=768 per row -> bf16.  row=(b,n); src row at b*sb+tOff+n*D
__global__ __launch_bounds__(256)
void ln_bf16_kernel(unsigned short* out, const float* src,
                    const float* gamma, const float* beta, long sb, long tOff) {
  __shared__ float red[256];
  int row = blockIdx.x;
  int b = row >> 10, n = row & 1023;
  const float* p = src + (long)b * sb + tOff + (long)n * Dc;
  int tid = threadIdx.x;
  float x0 = p[tid], x1 = p[tid + 256], x2 = p[tid + 512];
  red[tid] = x0 + x1 + x2;
  __syncthreads();
  for (int o = 128; o > 0; o >>= 1) { if (tid < o) red[tid] += red[tid + o]; __syncthreads(); }
  float mean = red[0] * (1.f / (float)Dc);
  __syncthreads();
  float d0 = x0 - mean, d1 = x1 - mean, d2 = x2 - mean;
  red[tid] = d0 * d0 + d1 * d1 + d2 * d2;
  __syncthreads();
  for (int o = 128; o > 0; o >>= 1) { if (tid < o) red[tid] += red[tid + o]; __syncthreads(); }
  float rstd = rsqrtf(red[0] * (1.f / (float)Dc) + 1e-5f);
  long ro = (long)row * Dc;
  out[ro + tid      ] = f2bf(d0 * rstd * gamma[tid      ] + beta[tid      ]);
  out[ro + tid + 256] = f2bf(d1 * rstd * gamma[tid + 256] + beta[tid + 256]);
  out[ro + tid + 512] = f2bf(d2 * rstd * gamma[tid + 512] + beta[tid + 512]);
}

// split qkv (2048 x 2304 f32) -> per-head bf16 q/k (elu+1) and v, layout (B,H,N,64)
__global__ void qkv_epilogue_kernel(const float* qkv, unsigned short* qb,
                                    unsigned short* kb, unsigned short* vb) {
  long i = (long)blockIdx.x * blockDim.x + threadIdx.x;
  if (i >= (long)ROWS * TDc) return;
  int col = (int)(i % TDc);
  int row = (int)(i / TDc);
  int b = row >> 10, n = row & 1023;
  int which = col / Dc;
  int c = col - which * Dc;
  int h = c >> 6, d = c & 63;
  long o = (((long)(b * Hc + h)) * Nc + n) * 64 + d;
  float v = qkv[i];
  if      (which == 0) qb[o] = f2bf(elu1(v));
  else if (which == 1) kb[o] = f2bf(elu1(v));
  else                 vb[o] = f2bf(v);
}

// x2 = attn + X[:,t]; hnext = attn + hin
__global__ void residual_kernel(float* x2, float* hnext, const float* attn,
                                const float* X, const float* hin, int t) {
  long i = (long)blockIdx.x * blockDim.x + threadIdx.x;
  if (i >= BNDl) return;
  long b = i / NDl, rem = i % NDl;
  float a = attn[i];
  x2[i]    = a + X[b * LNDl + (long)t * NDl + rem];
  hnext[i] = a + hin[i];
}

// Y[:,t] (=|+=) x2 + y
__global__ void finalize_kernel(float* Y, const float* x2, const float* y,
                                int t, int addFlag) {
  long i = (long)blockIdx.x * blockDim.x + threadIdx.x;
  if (i >= BNDl) return;
  long b = i / NDl, rem = i % NDl;
  long o = b * LNDl + (long)t * NDl + rem;
  float v = x2[i] + y[i];
  Y[o] = addFlag ? (Y[o] + v) : v;
}

// ---------------------------------------------------------------------------
// Host orchestration
// ---------------------------------------------------------------------------
extern "C" void kernel_launch(void* const* d_in, const int* in_sizes, int n_in,
                              void* d_out, int out_size, void* d_ws, size_t ws_size,
                              hipStream_t stream) {
  (void)in_sizes; (void)n_in; (void)out_size; (void)ws_size;

  const float* x      = (const float*)d_in[0];
  const float* hidden = (const float*)d_in[1];
  const float* spos   = (const float*)d_in[2];
  const float* tpos   = (const float*)d_in[3];
  const float* lni_g  = (const float*)d_in[4];
  const float* lni_b  = (const float*)d_in[5];
  const float* lnh_g  = (const float*)d_in[6];
  const float* lnh_b  = (const float*)d_in[7];
  const float* lno_g  = (const float*)d_in[8];
  const float* lno_b  = (const float*)d_in[9];
  const float* Wqkv   = (const float*)d_in[10];
  const float* bqkv   = (const float*)d_in[11];
  const float* Wqkvh  = (const float*)d_in[12];
  const float* bqkvh  = (const float*)d_in[13];
  const float* Wout   = (const float*)d_in[14];
  const float* bout   = (const float*)d_in[15];
  const float* W1     = (const float*)d_in[16];
  const float* b1     = (const float*)d_in[17];
  const float* W2     = (const float*)d_in[18];
  const float* b2     = (const float*)d_in[19];
  float* out = (float*)d_out;

  // bump allocator over workspace
  char* base = (char*)d_ws; size_t off = 0;
  auto alloc = [&](size_t bytes) -> void* {
    off = (off + 255) & ~(size_t)255;
    void* p = base + off; off += bytes; return p;
  };

  const long nWqkv = 4L * Dc * TDc, nWout = 4L * Dc * Dc, nW1 = 4L * Dc * Mc;
  unsigned short* Wqkv_bf  = (unsigned short*)alloc(2 * nWqkv);
  unsigned short* Wqkvh_bf = (unsigned short*)alloc(2 * nWqkv);
  unsigned short* Wout_bf  = (unsigned short*)alloc(2 * nWout);
  unsigned short* W1_bf    = (unsigned short*)alloc(2 * nW1);
  unsigned short* W2_bf    = (unsigned short*)alloc(2 * nW1);

  float*          Xw    = (float*)alloc(4 * BLNDl);
  float*          Ysum  = (float*)alloc(4 * BLNDl);
  float*          hA    = (float*)alloc(4 * BNDl);
  float*          hB    = (float*)alloc(4 * BNDl);
  float*          hin   = (float*)alloc(4 * BNDl);
  unsigned short* xn_bf = (unsigned short*)alloc(2 * BNDl);
  unsigned short* hn_bf = (unsigned short*)alloc(2 * BNDl);
  float*          qkv_f = (float*)alloc(4L * ROWS * TDc);
  unsigned short* q_bf  = (unsigned short*)alloc(2 * BNDl);
  unsigned short* k_bf  = (unsigned short*)alloc(2 * BNDl);
  unsigned short* v_bf  = (unsigned short*)alloc(2 * BNDl);
  unsigned short* kv_bf = (unsigned short*)alloc(2L * Bc * Hc * 64 * 64);
  unsigned short* o_bf  = (unsigned short*)alloc(2 * BNDl);
  float*          attn_f= (float*)alloc(4 * BNDl);
  float*          x2_f  = (float*)alloc(4 * BNDl);
  unsigned short* x2n_bf= (unsigned short*)alloc(2 * BNDl);
  unsigned short* y1_bf = (unsigned short*)alloc(2L * ROWS * Mc);
  float*          y_f   = (float*)alloc(4 * BNDl);

  auto conv = [&](unsigned short* dst, const float* src, long n) {
    f32_to_bf16_kernel<<<(int)((n + 255) / 256), 256, 0, stream>>>(dst, src, n);
  };
  conv(Wqkv_bf,  Wqkv,  nWqkv);
  conv(Wqkvh_bf, Wqkvh, nWqkv);
  conv(Wout_bf,  Wout,  nWout);
  conv(W1_bf,    W1,    nW1);
  conv(W2_bf,    W2,    nW1);

  auto gemm = [&](const unsigned short* A, const unsigned short* Bw,
                  float* C, unsigned short* Cbf, const float* bias,
                  int M, int N, int K,
                  long sAm, long sAk, long sBk, long sBn, long sCm, long sCbm,
                  long aBO, long aBI, long bBO, long bBI, long cBO, long cBI,
                  int hDiv, int batch, int accF, int act) {
    GemmArgs ga;
    ga.A = A; ga.Bm = Bw; ga.C = C; ga.Cbf = Cbf; ga.bias = bias;
    ga.M = M; ga.N = N; ga.K = K;
    ga.sAm = sAm; ga.sAk = sAk; ga.sBk = sBk; ga.sBn = sBn;
    ga.sCm = sCm; ga.sCbm = sCbm;
    ga.aBO = aBO; ga.aBI = aBI; ga.bBO = bBO; ga.bBI = bBI;
    ga.cBO = cBO; ga.cBI = cBI;
    ga.hDiv = hDiv > 0 ? hDiv : 1; ga.accumulate = accF; ga.act = act;
    dim3 grid((N + 63) / 64, (M + 127) / 128, batch);
    gemm_bf16_wmma<<<grid, dim3(256), 0, stream>>>(ga);
  };

  const int gBND  = (int)((BNDl + 255) / 256);
  const int gBLND = (int)((BLNDl + 255) / 256);
  const int gQKV  = (int)(((long)ROWS * TDc + 255) / 256);

  const float* xsrc = x;
  for (int layer = 0; layer < 2; ++layer) {
    // x = x + pos
    add_pos_kernel<<<gBLND, 256, 0, stream>>>(Xw, xsrc, tpos, spos);
    float* ytgt = (layer == 0) ? Ysum : out;

    for (int dir = 0; dir < 2; ++dir) {   // 0 = forward scan, 1 = backward scan
      (void)hipMemcpyAsync(hA, hidden, sizeof(float) * BNDl,
                           hipMemcpyDeviceToDevice, stream);
      float* hcur = hA; float* hnxt = hB;

      const int pidx = dir * 2 + layer;   // params shaped (2, NUM_LAYERS, ...)
      const unsigned short* wqkv_s  = Wqkv_bf  + (long)pidx * Dc * TDc;
      const unsigned short* wqkvh_s = Wqkvh_bf + (long)pidx * Dc * TDc;
      const unsigned short* wout_s  = Wout_bf  + (long)pidx * Dc * Dc;
      const unsigned short* w1_s    = W1_bf    + (long)pidx * Dc * Mc;
      const unsigned short* w2_s    = W2_bf    + (long)pidx * Mc * Dc;
      const float* bqkv_s  = bqkv  + (long)pidx * TDc;
      const float* bqkvh_s = bqkvh + (long)pidx * TDc;
      const float* bout_s  = bout  + (long)pidx * Dc;
      const float* b1_s    = b1    + (long)pidx * Mc;
      const float* b2_s    = b2    + (long)pidx * Dc;

      for (int s = 0; s < Lc; ++s) {
        const int t  = (dir == 0) ? s : (Lc - 1 - s);
        const int pt = (dir == 0) ? layer : t;   // reference: fwd uses pos[:, layer]

        hin_kernel<<<gBND, 256, 0, stream>>>(hin, hcur, tpos, spos, pt);
        ln_bf16_kernel<<<ROWS, 256, 0, stream>>>(xn_bf, Xw,
            lni_g + (long)pidx * Dc, lni_b + (long)pidx * Dc, LNDl, (long)t * NDl);
        ln_bf16_kernel<<<ROWS, 256, 0, stream>>>(hn_bf, hin,
            lnh_g + (long)pidx * Dc, lnh_b + (long)pidx * Dc, NDl, 0L);

        // qkv = xn@Wqkv + bqkv + hn@Wqkvh + bqkvh
        gemm(xn_bf, wqkv_s,  qkv_f, nullptr, bqkv_s,  ROWS, TDc, Dc,
             Dc, 1, TDc, 1, TDc, 0, 0,0,0,0,0,0, 1, 1, 0, 0);
        gemm(hn_bf, wqkvh_s, qkv_f, nullptr, bqkvh_s, ROWS, TDc, Dc,
             Dc, 1, TDc, 1, TDc, 0, 0,0,0,0,0,0, 1, 1, 1, 0);
        qkv_epilogue_kernel<<<gQKV, 256, 0, stream>>>(qkv_f, q_bf, k_bf, v_bf);

        // kv[bh] = k^T @ v (64x64, K=N=1024), batched over 24 heads; bf16 out
        gemm(k_bf, v_bf, nullptr, kv_bf, nullptr, 64, 64, Nc,
             1, 64, 64, 1, 0, 64,
             65536, 0, 65536, 0, 4096, 0, 1, Bc * Hc, 0, 0);

        // o[bh] = q @ kv, written bf16 directly into merged (B,N,768) layout
        gemm(q_bf, kv_bf, nullptr, o_bf, nullptr, Nc, 64, 64,
             64, 1, 64, 1, 0, Dc,
             (long)Hc * 65536, 65536, (long)Hc * 4096, 4096,
             (long)Nc * Dc, 64, Hc, Bc * Hc, 0, 0);

        // attn = o @ Wout + bout
        gemm(o_bf, wout_s, attn_f, nullptr, bout_s, ROWS, Dc, Dc,
             Dc, 1, Dc, 1, Dc, 0, 0,0,0,0,0,0, 1, 1, 0, 0);

        residual_kernel<<<gBND, 256, 0, stream>>>(x2_f, hnxt, attn_f, Xw, hin, t);
        ln_bf16_kernel<<<ROWS, 256, 0, stream>>>(x2n_bf, x2_f,
            lno_g + (long)pidx * Dc, lno_b + (long)pidx * Dc, NDl, 0L);

        // MLP: y1 = gelu(x2n@W1 + b1) (bf16 epilogue), y = y1@W2 + b2
        gemm(x2n_bf, w1_s, nullptr, y1_bf, b1_s, ROWS, Mc, Dc,
             Dc, 1, Mc, 1, 0, Mc, 0,0,0,0,0,0, 1, 1, 0, 1);
        gemm(y1_bf, w2_s, y_f, nullptr, b2_s, ROWS, Dc, Mc,
             Mc, 1, Dc, 1, Dc, 0, 0,0,0,0,0,0, 1, 1, 0, 0);

        finalize_kernel<<<gBND, 256, 0, stream>>>(ytgt, x2_f, y_f, t, dir);

        float* tmp = hcur; hcur = hnxt; hnxt = tmp;
      }
    }
    xsrc = Ysum;   // layer-1 input is layer-0 output (layer-1 writes d_out)
  }
}
